// NeuralAstar_10144712753348
// MI455X (gfx1250) — matrix operand order repaired and checked by reading, not compiled, not run on hardware
//
#include <hip/hip_runtime.h>

typedef __attribute__((ext_vector_type(16))) _Float16 v16h;
typedef __attribute__((ext_vector_type(8)))  float    v8f;

// ---------------------------------------------------------------------------
// WMMA 3x3 SAME conv as GEMM:  out[b, n, pix] = relu( sum_k A[pix,k]*W[k,n] + bias )
//   A = on-the-fly im2col (M = H*W pixels, K = CIN*9), W = [Cout][Cin][3][3] f32
// One wave per 16x16 (pixels x out-channels) tile; K consumed in chunks of 32
// via v_wmma_f32_16x16x32_f16 with f32 accumulation.
// MODE 0: encoder input  (ch0 = map, ch1 = start+goal), f32 sources
// MODE 1: generic f16 activation tensor [B][CIN][H][W]
// MODE 2: decoder concat (ci<64 -> nearest-upsampled e2 [B][64][16][16],
//                         ci>=64 -> e1 [B][32][32][32])
// ---------------------------------------------------------------------------
template<int CIN, int COUT, int HH, int WW, int MODE>
__global__ __launch_bounds__(32)
void conv3x3_wmma(const float* __restrict__ f0, const float* __restrict__ f1,
                  const float* __restrict__ f2,
                  const _Float16* __restrict__ hin,
                  const _Float16* __restrict__ he2,
                  const _Float16* __restrict__ he1,
                  const float* __restrict__ wgt, const float* __restrict__ bias,
                  _Float16* __restrict__ out)
{
  constexpr int HW = HH * WW;
  constexpr int K  = CIN * 9;
  constexpr int KC = (K + 31) / 32;

  const int lane  = threadIdx.x;
  const int mBase = blockIdx.x * 16;   // pixel tile
  const int nBase = blockIdx.y * 16;   // out-channel tile
  const int b     = blockIdx.z;
  const int mrow  = lane & 15;
  const int hi    = lane >> 4;

  const int pix = mBase + mrow;
  const int py  = pix / WW, px = pix % WW;

  v8f acc = {};
  for (int kc = 0; kc < KC; ++kc) {
    // ---- A fragment: 16x32 f16. Lane holds row m=lane&15; half-wave splits K
    //      (VGPR v<4: K = 8*hi + 2v + h ; v>=4: K = 16 + 8*hi + 2(v-4) + h) ----
    v16h a;
#pragma unroll
    for (int e = 0; e < 16; ++e) {
      const int v  = e >> 1, hb = e & 1;
      const int kin = (v < 4) ? (hi * 8 + 2 * v + hb)
                              : (16 + hi * 8 + 2 * (v - 4) + hb);
      const int k = kc * 32 + kin;
      float val = 0.0f;
      if (k < K) {
        const int ci = k / 9, r = k - ci * 9;
        const int y = py + r / 3 - 1;
        const int x = px + r % 3 - 1;
        if (y >= 0 && y < HH && x >= 0 && x < WW) {
          if (MODE == 0) {
            const int o = b * HW + y * WW + x;
            val = (ci == 0) ? f0[o] : (f1[o] + f2[o]);
          } else if (MODE == 2) {
            if (ci < 64)
              val = (float)he2[((b * 64 + ci) << 8) + ((y >> 1) << 4) + (x >> 1)];
            else
              val = (float)he1[((b * 32 + (ci - 64)) << 10) + (y << 5) + x];
          } else {
            val = (float)hin[(b * CIN + ci) * HW + y * WW + x];
          }
        }
      }
      a[e] = (_Float16)val;
    }
    // ---- B fragment: 32x16 f16. Lane holds column n = lane&15;
    //      elements hold K in the same order as the A fragment ----
    v16h bf;
    const int ncol = nBase + (lane & 15);
#pragma unroll
    for (int e = 0; e < 16; ++e) {
      const int kin = (e < 8) ? (hi * 8 + e) : (16 + hi * 8 + (e - 8));
      const int krow = kc * 32 + kin;
      const int ci = krow / 9, r = krow - ci * 9;
      float val = 0.0f;
      if (krow < K) val = wgt[(ncol * CIN + ci) * 9 + r];
      bf[e] = (_Float16)val;
    }
    acc = __builtin_amdgcn_wmma_f32_16x16x32_f16(false, a, false, bf,
                                                 (short)0, acc, false, false);
  }
  // ---- epilogue: C/D layout lane 0-15 -> N=lane, VGPR j -> M = 8*hi + j ----
  const int n  = nBase + (lane & 15);
  const int m0 = mBase + hi * 8;
#pragma unroll
  for (int j = 0; j < 8; ++j) {
    float o = acc[j] + bias[n];
    o = fmaxf(o, 0.0f);
    out[(b * COUT + n) * HW + m0 + j] = (_Float16)o;
  }
}

// ---------------------------------------------------------------------------
// 2x2 max-pool on e1 [32,32,32,32] -> [32,32,16,16]
// ---------------------------------------------------------------------------
__global__ __launch_bounds__(256)
void maxpool2(const _Float16* __restrict__ in, _Float16* __restrict__ out)
{
  const int gid = blockIdx.x * blockDim.x + threadIdx.x;
  if (gid >= 32 * 32 * 16 * 16) return;
  const int px = gid & 15, py = (gid >> 4) & 15, c = (gid >> 8) & 31, b = gid >> 13;
  const _Float16* base = in + ((size_t)(b * 32 + c) << 10) + (py * 2) * 32 + px * 2;
  float m = fmaxf(fmaxf((float)base[0], (float)base[1]),
                  fmaxf((float)base[32], (float)base[33]));
  out[gid] = (_Float16)m;
}

// ---------------------------------------------------------------------------
// 1x1 conv (N=1) + sigmoid -> pred_cost_maps (f32, written straight into d_out)
// ---------------------------------------------------------------------------
__global__ __launch_bounds__(256)
void conv1x1_sigmoid(const _Float16* __restrict__ dd, const float* __restrict__ wo,
                     const float* __restrict__ bo, float* __restrict__ outCost)
{
  const int gid = blockIdx.x * blockDim.x + threadIdx.x;
  if (gid >= 32 * 1024) return;
  const int b = gid >> 10, pix = gid & 1023;
  float s = bo[0];
#pragma unroll
  for (int ci = 0; ci < 32; ++ci)
    s += (float)dd[((size_t)(b * 32 + ci) << 10) + pix] * wo[ci];
  outCost[gid] = 1.0f / (1.0f + __expf(-s));
}

// ---------------------------------------------------------------------------
// Differentiable A* forward: one wave32 per batch element, all state in LDS.
// lane = grid row; argmax via packed-u64 shfl reduction (exact tie-break =
// lowest flat index, matching jnp.argmax). 256 sequential steps + backtrack.
// ---------------------------------------------------------------------------
__device__ inline unsigned long long wave_max_u64(unsigned long long v)
{
  for (int off = 16; off > 0; off >>= 1) {
    unsigned lo = __shfl_xor((unsigned)v, off, 32);
    unsigned hi = __shfl_xor((unsigned)(v >> 32), off, 32);
    unsigned long long o = ((unsigned long long)hi << 32) | lo;
    v = (o > v) ? o : v;
  }
  return v;
}

__global__ __launch_bounds__(32)
void astar_kernel(const float* __restrict__ map, const float* __restrict__ start,
                  const float* __restrict__ goal, const float* __restrict__ cost,
                  float* __restrict__ outHist, float* __restrict__ outPath)
{
  const int b    = blockIdx.x;
  const int lane = threadIdx.x;             // owns row y = lane
  __shared__ float s_cost[1024], s_hmap[1024], s_g[1024];
  __shared__ float s_open[1024], s_hist[1024], s_obst[1024], s_path[1024];
  __shared__ short s_par[1024];

  const size_t base = (size_t)b * 1024;

  // ---- init: load maps, find goal index ----
  int gidx = -1;
  for (int x = 0; x < 32; ++x) {
    const int cell = lane * 32 + x;
    s_obst[cell] = map[base + cell];
    s_cost[cell] = cost[base + cell];
    s_open[cell] = start[base + cell];
    s_hist[cell] = 0.0f;
    s_g[cell]    = 0.0f;
    if (goal[base + cell] > 0.5f) gidx = cell;
  }
  for (int off = 16; off > 0; off >>= 1) {
    int o = __shfl_xor(gidx, off, 32);
    gidx = (o > gidx) ? o : gidx;
  }
  const int gy = gidx >> 5, gx = gidx & 31;

  // ---- h = chebyshev + 0.001*euclid + cost ; parents = goal ; path = goal ----
  for (int x = 0; x < 32; ++x) {
    const int cell = lane * 32 + x;
    int dy = lane - gy; dy = dy < 0 ? -dy : dy;
    int dx = x - gx;    dx = dx < 0 ? -dx : dx;
    const int mn = dy < dx ? dy : dx;
    const float cheb = (float)(dy + dx - mn);
    const float euc  = sqrtf((float)(dy * dy + dx * dx));
    s_hmap[cell] = cheb + 0.001f * euc + s_cost[cell];
    s_par[cell]  = (short)gidx;
    s_path[cell] = (cell == gidx) ? 1.0f : 0.0f;
  }
  __syncthreads();   // single-wave WG: lowers to waitcnt + S_NOP

  const float kInvSqrtW = 0.17677669529663687f;   // 1/sqrt(32)
  for (int t = 0; t < 256; ++t) {
    // ---- selection: argmax of exp(-f/sqrt(w)) * open, tie -> lowest index ----
    unsigned long long key = 0ull;
    for (int x = 0; x < 32; ++x) {
      const int cell = lane * 32 + x;
      const float f  = 0.5f * (s_g[cell] + s_hmap[cell]);   // G_RATIO = 0.5
      const float fe = __expf(-f * kInvSqrtW) * s_open[cell];
      const unsigned long long k =
          ((unsigned long long)__float_as_uint(fe) << 32) | (unsigned)(~cell);
      key = (k > key) ? k : key;
    }
    key = wave_max_u64(key);
    const int sel = (int)(~(unsigned)key);
    const int sy = sel >> 5, sx = sel & 31;

    const float gpc = s_g[sel] + s_cost[sel];     // g2 value pushed to neighbors
    const bool unsolved = (sel != gidx);

    if (lane == sy) {
      s_hist[sel] = 1.0f;                          // histories |= selected
      if (unsolved) s_open[sel] = 0.0f;            // open -= is_unsolved*selected
    }
    __syncthreads();

    // ---- expand: 8-neighbors; idx = obst & !open & !hist (sel never its own
    //      neighbor, so ordering matches the reference exactly) ----
    if (lane >= sy - 1 && lane <= sy + 1) {
      const int xlo = sx > 0 ? sx - 1 : 0;
      const int xhi = sx < 31 ? sx + 1 : 31;
      for (int x = xlo; x <= xhi; ++x) {
        const int cell = lane * 32 + x;
        if (cell == sel) continue;
        if (s_obst[cell] > 0.5f && s_open[cell] < 0.5f && s_hist[cell] < 0.5f) {
          s_g[cell]    = gpc;
          s_open[cell] = 1.0f;
          s_par[cell]  = (short)sel;
        }
      }
    }
    __syncthreads();
  }

  // ---- outputs: histories, then T-step backtrack of parents ----
  for (int x = 0; x < 32; ++x) {
    const int cell = lane * 32 + x;
    outHist[base + cell] = s_hist[cell];
  }
  if (lane == 0) {
    int loc = (int)s_par[gidx];
    for (int t = 0; t < 256; ++t) {
      s_path[loc] = 1.0f;
      loc = (int)s_par[loc];
    }
  }
  __syncthreads();
  for (int x = 0; x < 32; ++x) {
    const int cell = lane * 32 + x;
    outPath[base + cell] = s_path[cell];
  }
}

// ---------------------------------------------------------------------------
extern "C" void kernel_launch(void* const* d_in, const int* in_sizes, int n_in,
                              void* d_out, int out_size, void* d_ws, size_t ws_size,
                              hipStream_t stream)
{
  (void)in_sizes; (void)n_in; (void)out_size; (void)ws_size;
  const float* map   = (const float*)d_in[0];
  const float* start = (const float*)d_in[1];
  const float* goal  = (const float*)d_in[2];
  const float* w1 = (const float*)d_in[3];  const float* b1 = (const float*)d_in[4];
  const float* w2 = (const float*)d_in[5];  const float* b2 = (const float*)d_in[6];
  const float* w3 = (const float*)d_in[7];  const float* b3 = (const float*)d_in[8];
  const float* w4 = (const float*)d_in[9];  const float* b4 = (const float*)d_in[10];
  const float* wd = (const float*)d_in[11]; const float* bd = (const float*)d_in[12];
  const float* wo = (const float*)d_in[13]; const float* bo = (const float*)d_in[14];

  // f16 activation scratch (≈8.5 MB total)
  _Float16* t0 = (_Float16*)d_ws;          // conv1 out  [32,32,32,32]
  _Float16* e1 = t0 + (32 * 32 * 1024);    // conv2 out  [32,32,32,32]
  _Float16* pp = e1 + (32 * 32 * 1024);    // pool       [32,32,16,16]
  _Float16* t1 = pp + (32 * 32 * 256);     // conv3 out  [32,64,16,16]
  _Float16* e2 = t1 + (32 * 64 * 256);     // conv4 out  [32,64,16,16]
  _Float16* dd = e2 + (32 * 64 * 256);     // convd out  [32,32,32,32]

  float* outHist = (float*)d_out;              // [32,1,32,32]
  float* outPath = outHist + 32 * 1024;        // [32,1,32,32]
  float* outCost = outHist + 2 * 32 * 1024;    // [32,1,32,32]

  const dim3 wv(32);
  conv3x3_wmma<2, 32, 32, 32, 0><<<dim3(64, 2, 32), wv, 0, stream>>>(
      map, start, goal, nullptr, nullptr, nullptr, w1, b1, t0);
  conv3x3_wmma<32, 32, 32, 32, 1><<<dim3(64, 2, 32), wv, 0, stream>>>(
      nullptr, nullptr, nullptr, t0, nullptr, nullptr, w2, b2, e1);
  maxpool2<<<dim3((32 * 32 * 16 * 16 + 255) / 256), dim3(256), 0, stream>>>(e1, pp);
  conv3x3_wmma<32, 64, 16, 16, 1><<<dim3(16, 4, 32), wv, 0, stream>>>(
      nullptr, nullptr, nullptr, pp, nullptr, nullptr, w3, b3, t1);
  conv3x3_wmma<64, 64, 16, 16, 1><<<dim3(16, 4, 32), wv, 0, stream>>>(
      nullptr, nullptr, nullptr, t1, nullptr, nullptr, w4, b4, e2);
  conv3x3_wmma<96, 32, 32, 32, 2><<<dim3(64, 2, 32), wv, 0, stream>>>(
      nullptr, nullptr, nullptr, nullptr, e2, e1, wd, bd, dd);
  conv1x1_sigmoid<<<dim3(32 * 1024 / 256), dim3(256), 0, stream>>>(dd, wo, bo, outCost);
  astar_kernel<<<dim3(32), wv, 0, stream>>>(map, start, goal, outCost, outHist, outPath);
}
